// LinearREncoder_38087769981504
// MI455X (gfx1250) — matrix operand order repaired and checked by reading, not compiled, not run on hardware
//
#include <hip/hip_runtime.h>
#include <stdint.h>

// ---------------------------------------------------------------------------
// LinearREncoder for MI455X (gfx1250, wave32, WMMA + Tensor Data Mover).
//   r_aggr[b] = maskedmean_n( relu(relu([x|y]W1+b1)W2+b2) ) W3 + b3
// Layer 3 is linear -> folded AFTER the masked mean (275 GF -> 0.5 GF).
// Big GEMMs: bf16 v_wmma_f32_16x16x32_bf16, fp32 accumulate.
// Tile staging: TDM tensor_load_to_lds (no VGPR staging -> no scratch spills),
// double-buffered LDS, TENSORcnt-synchronized.
// ---------------------------------------------------------------------------

typedef __bf16 bf16_t;
typedef bf16_t v16bf __attribute__((ext_vector_type(16)));
typedef float  v8f   __attribute__((ext_vector_type(8)));

#define BSZ   64
#define NPT   512
#define XD    128
#define YD    128
#define DIN   256
#define HD    4096
#define RD    1024
#define MROWS (BSZ * NPT)   // 32768

#define BM  128
#define BN  128
#define BK  64
#define LDT 72          // LDS row stride in bf16 elems (= BK + 8 pad; 144B rows)

static_assert(MROWS % BM == 0 && HD % BN == 0 && DIN % BK == 0 && HD % BK == 0, "tiling");

#if __has_builtin(__builtin_amdgcn_tensor_load_to_lds)
#define USE_TDM 1
#else
#define USE_TDM 0
#endif

// ---- bf16 helpers (storage type = unsigned short) --------------------------
static __device__ __forceinline__ unsigned short f32_to_bf16(float f) {
    union { float f; unsigned u; } c; c.f = f;
    unsigned u = c.u;
    if ((u & 0x7F800000u) == 0x7F800000u && (u & 0x007FFFFFu)) // NaN
        return (unsigned short)((u >> 16) | 1u);
    return (unsigned short)((u + 0x7FFFu + ((u >> 16) & 1u)) >> 16); // RNE
}
static __device__ __forceinline__ float bf16_to_f32(unsigned short h) {
    union { unsigned u; float f; } c; c.u = ((unsigned)h) << 16; return c.f;
}

// LDS aperture: generic shared pointer carries the LDS byte offset in addr[31:0].
static __device__ __forceinline__ unsigned lds_byte_off(const void* p) {
    return (unsigned)(unsigned long long)(uintptr_t)p;
}

#if USE_TDM
typedef unsigned int tdm_v4u __attribute__((ext_vector_type(4)));
typedef int          tdm_v4i __attribute__((ext_vector_type(4)));
typedef int          tdm_v8i __attribute__((ext_vector_type(8)));

// Issue one TDM 2-D tile load: tile_d0 x tile_d1 bf16 elements, row stride
// `stride0` elements in memory, into LDS at lds_addr with +16B padding after
// every 128B row (pad_interval code 4 = 32 DWORDs, pad_amount code 3 = 4 DWORDs)
// -> reproduces the LDT=72 LDS row stride.
static __device__ __forceinline__ void tdm_load_tile_2d(
    unsigned lds_addr, const void* gptr,
    unsigned tile_d0, unsigned tile_d1,
    unsigned tensor_d0, unsigned tensor_d1, unsigned long long stride0)
{
    unsigned long long ga = (unsigned long long)(uintptr_t)gptr;
    tdm_v4u g0;
    g0[0] = 1u;                                           // count=1, user D#
    g0[1] = lds_addr;                                     // LDS byte address
    g0[2] = (unsigned)(ga & 0xFFFFFFFFull);               // global_addr[31:0]
    g0[3] = (unsigned)((ga >> 32) & 0x01FFFFFFull) | (2u << 30); // [56:32] | type=2

    tdm_v8i g1;
    g1[0] = (int)((1u << 16) | (1u << 20) | (4u << 22) | (3u << 25));
            // data_size=2B | pad_enable | pad_interval=32dw | pad_amount=4dw
    g1[1] = (int)((tensor_d0 & 0xFFFFu) << 16);           // tensor_dim0[15:0]
    g1[2] = (int)((tensor_d0 >> 16) | ((tensor_d1 & 0xFFFFu) << 16));
    g1[3] = (int)((tensor_d1 >> 16) | (tile_d0 << 16));   // tile_dim0
    g1[4] = (int)(tile_d1 & 0xFFFFu);                     // tile_dim1 (tile_dim2=0)
    g1[5] = (int)(stride0 & 0xFFFFFFFFull);               // tensor_dim0_stride
    g1[6] = (int)((stride0 >> 32) & 0xFFFFull);           // (dim1_stride=0)
    g1[7] = 0;

    tdm_v4i z4 = {0, 0, 0, 0};
#if __has_include(<hip/amd_detail/amd_gfx1250_TDM.h>)
    tdm_v8i z8 = {0, 0, 0, 0, 0, 0, 0, 0};
    __builtin_amdgcn_tensor_load_to_lds(g0, g1, z4, z4, z8, 0);
#else
    __builtin_amdgcn_tensor_load_to_lds(g0, g1, z4, z4, 0);
#endif
}
#endif // USE_TDM

// ---- 1) concat x|y and convert to bf16: A[32768, 256] ----------------------
__global__ __launch_bounds__(256) void k_concat_bf16(
    const float* __restrict__ x, const float* __restrict__ y,
    unsigned short* __restrict__ A)
{
    int idx = blockIdx.x * 256 + threadIdx.x;
    int row = idx >> 8;
    int d   = idx & 255;
    float v = (d < XD) ? x[row * XD + d] : y[row * YD + (d - XD)];
    A[idx] = f32_to_bf16(v);
}

// ---- 2) transpose + convert weights: Wt[n*K + k] = bf16(W[k*N + n]) --------
__global__ __launch_bounds__(256) void k_transpose_bf16(
    const float* __restrict__ W, unsigned short* __restrict__ Wt, int K, int N)
{
    long long idx = (long long)blockIdx.x * 256 + threadIdx.x;
    int n = (int)(idx / K);
    int k = (int)(idx % K);
    Wt[idx] = f32_to_bf16(W[(long long)k * N + n]);
}

// ---- 3/4) C = act(A @ B + bias), A:[M,K] bf16 rm, Bt:[N,K] bf16 rm ---------
// Block = 256 threads = 8 waves; block tile 128x128; wave tile 32x64
// (2x4 WMMA 16x16 accumulators).  K chunked by 64.  Tiles staged into LDS by
// the TDM (waves 0/1 issue, TENSORcnt-waited), double-buffered.
__global__ __launch_bounds__(256) void k_wmma_gemm_bias_act(
    const unsigned short* __restrict__ A,
    const unsigned short* __restrict__ Bt,
    const float* __restrict__ bias,
    unsigned short* __restrict__ C,
    int M, int N, int K, int do_relu)
{
    __shared__ unsigned short sA[2][BM][LDT];
    __shared__ unsigned short sB[2][BN][LDT];

    const int t    = threadIdx.x;
    const int lane = t & 31;
    const int wave = t >> 5;           // 0..7
    const int half = (lane >> 4) & 1;  // lanes 0-15 vs 16-31
    const int l16  = lane & 15;

    const int m0 = blockIdx.y * BM;
    const int n0 = blockIdx.x * BN;

    const int wm0 = (wave & 3) * 32;   // 4 waves along M
    const int wn0 = (wave >> 2) * 64;  // 2 waves along N

    v8f acc[2][4] = {};
    const int nk = K / BK;

    auto compute = [&](int pb) {
        #pragma unroll
        for (int kk = 0; kk < BK; kk += 32) {
            // A frag (16x32): lane l16 = row; e0..7 -> K = kk+half*8+e,
            // e8..15 -> K = kk+16+half*8+(e-8).  Two ds_load_b128.
            v16bf af[2];
            #pragma unroll
            for (int mi = 0; mi < 2; ++mi) {
                const unsigned short* p = &sA[pb][wm0 + mi * 16 + l16][kk + half * 8];
                union { uint4 q[2]; v16bf v; } u;
                u.q[0] = *reinterpret_cast<const uint4*>(p);
                u.q[1] = *reinterpret_cast<const uint4*>(p + 16);
                af[mi] = u.v;
            }
            // B frag (32x16): lane l16 = col; e0..7 -> K = kk+half*8+e,
            // e8..15 -> K = kk+16+half*8+(e-8).
            v16bf bfg[4];
            #pragma unroll
            for (int ni = 0; ni < 4; ++ni) {
                const unsigned short* p = &sB[pb][wn0 + ni * 16 + l16][kk + half * 8];
                union { uint4 q[2]; v16bf v; } u;
                u.q[0] = *reinterpret_cast<const uint4*>(p);
                u.q[1] = *reinterpret_cast<const uint4*>(p + 16);
                bfg[ni] = u.v;
            }
            #pragma unroll
            for (int mi = 0; mi < 2; ++mi)
                #pragma unroll
                for (int ni = 0; ni < 4; ++ni)
                    acc[mi][ni] = __builtin_amdgcn_wmma_f32_16x16x32_bf16(
                        false, af[mi], false, bfg[ni],
                        (short)0, acc[mi][ni], false, false);
        }
    };

#if USE_TDM
    auto issue = [&](int kc, int pb) {
        if (wave == 0) {
            tdm_load_tile_2d(lds_byte_off(&sA[pb][0][0]),
                             A + (size_t)m0 * K + (size_t)kc * BK,
                             BK, BM, (unsigned)K, (unsigned)M, (unsigned long long)K);
        } else if (wave == 1) {
            tdm_load_tile_2d(lds_byte_off(&sB[pb][0][0]),
                             Bt + (size_t)n0 * K + (size_t)kc * BK,
                             BK, BN, (unsigned)K, (unsigned)N, (unsigned long long)K);
        }
    };

    issue(0, 0);
    for (int kc = 0; kc < nk; ++kc) {
        const int pb = kc & 1;
        if (wave < 2) __builtin_amdgcn_s_wait_tensorcnt(0); // issuing wave's DMA done
        __syncthreads();                                    // tile visible to all
        if (kc + 1 < nk) issue(kc + 1, pb ^ 1);             // overlap next DMA w/ WMMA
        compute(pb);
        __syncthreads();                                    // all reads done before refill
    }
#else
    // Fallback: cooperative load/stage, transient registers only (no spills).
    for (int kc = 0; kc < nk; ++kc) {
        #pragma unroll
        for (int i = 0; i < 4; ++i) {
            int c = t + i * 256;
            int row = c >> 3, col = (c & 7) * 8;
            uint4 va = *reinterpret_cast<const uint4*>(A + (size_t)(m0 + row) * K + kc * BK + col);
            *reinterpret_cast<uint4*>(&sA[0][row][col]) = va;
            uint4 vb = *reinterpret_cast<const uint4*>(Bt + (size_t)(n0 + row) * K + kc * BK + col);
            *reinterpret_cast<uint4*>(&sB[0][row][col]) = vb;
        }
        __syncthreads();
        compute(0);
        __syncthreads();
    }
#endif

    // Epilogue: C/D 16x16 f32 layout: VGPR v -> row (v + 8*half), lane l16 -> col.
    #pragma unroll
    for (int ni = 0; ni < 4; ++ni) {
        int col = n0 + wn0 + ni * 16 + l16;
        float bv = bias[col];
        #pragma unroll
        for (int mi = 0; mi < 2; ++mi) {
            int rbase = m0 + wm0 + mi * 16 + half * 8;
            #pragma unroll
            for (int v = 0; v < 8; ++v) {
                float xv = acc[mi][ni][v] + bv;
                if (do_relu) xv = xv > 0.f ? xv : 0.f;
                C[(size_t)(rbase + v) * N + col] = f32_to_bf16(xv);
            }
        }
    }
}

// ---- 5) masked mean over tokens: g[b,k] = sum_n mask*h2 / max(cnt,1) -------
__global__ __launch_bounds__(256) void k_masked_mean(
    const unsigned short* __restrict__ h2, const unsigned char* __restrict__ mask,
    float* __restrict__ g)
{
    int k = blockIdx.x * 256 + threadIdx.x;  // 0..HD-1
    int b = blockIdx.y;
    float s = 0.f; int cnt = 0;
    for (int n = 0; n < NPT; ++n) {
        if (mask[b * NPT + n]) {
            s += bf16_to_f32(h2[(size_t)(b * NPT + n) * HD + k]);
            ++cnt;
        }
    }
    g[(size_t)b * HD + k] = s / (float)(cnt > 0 ? cnt : 1);
}

// ---- 6) out[b,r] = g[b,:] . W3[:,r] + b3[r]  (0.5 GF, plain fp32) ----------
__global__ __launch_bounds__(256) void k_final_gemv(
    const float* __restrict__ g, const float* __restrict__ W3,
    const float* __restrict__ b3, float* __restrict__ out)
{
    int r = blockIdx.x * 256 + threadIdx.x;  // 0..RD-1
    int b = blockIdx.y;
    const float* gp = g + (size_t)b * HD;
    float s = b3[r];
    for (int k = 0; k < HD; ++k) s += gp[k] * W3[(size_t)k * RD + r];
    out[(size_t)b * RD + r] = s;
}

// ---------------------------------------------------------------------------
extern "C" void kernel_launch(void* const* d_in, const int* in_sizes, int n_in,
                              void* d_out, int out_size, void* d_ws, size_t ws_size,
                              hipStream_t stream) {
    const float*         x    = (const float*)d_in[0];
    const float*         y    = (const float*)d_in[1];
    const unsigned char* mask = (const unsigned char*)d_in[2]; // jax bool = 1 byte
    const float*         W1   = (const float*)d_in[3];
    const float*         b1   = (const float*)d_in[4];
    const float*         W2   = (const float*)d_in[5];
    const float*         b2   = (const float*)d_in[6];
    const float*         W3   = (const float*)d_in[7];
    const float*         b3   = (const float*)d_in[8];
    float*               out  = (float*)d_out;

    // Workspace carve-up (~590 MB total).
    char* ws = (char*)d_ws;
    size_t off = 0;
    auto carve = [&](size_t bytes) -> void* {
        void* p = ws + off;
        off = (off + bytes + 255) & ~(size_t)255;
        return p;
    };
    unsigned short* Ain = (unsigned short*)carve((size_t)MROWS * DIN * 2);
    unsigned short* W1t = (unsigned short*)carve((size_t)HD * DIN * 2);
    unsigned short* W2t = (unsigned short*)carve((size_t)HD * HD * 2);
    unsigned short* h1  = (unsigned short*)carve((size_t)MROWS * HD * 2);
    unsigned short* h2  = (unsigned short*)carve((size_t)MROWS * HD * 2);
    float*          g   = (float*)carve((size_t)BSZ * HD * 4);
    (void)ws_size; (void)in_sizes; (void)n_in; (void)out_size;

    k_concat_bf16<<<(MROWS * DIN) / 256, 256, 0, stream>>>(x, y, Ain);
    k_transpose_bf16<<<(HD * DIN) / 256, 256, 0, stream>>>(W1, W1t, DIN, HD);
    k_transpose_bf16<<<((size_t)HD * HD) / 256, 256, 0, stream>>>(W2, W2t, HD, HD);
    k_wmma_gemm_bias_act<<<dim3(HD / BN, MROWS / BM), 256, 0, stream>>>(
        Ain, W1t, b1, h1, MROWS, HD, DIN, 1);
    k_wmma_gemm_bias_act<<<dim3(HD / BN, MROWS / BM), 256, 0, stream>>>(
        h1, W2t, b2, h2, MROWS, HD, HD, 1);
    k_masked_mean<<<dim3(HD / 256, BSZ), 256, 0, stream>>>(h2, mask, g);
    k_final_gemv<<<dim3(RD / 256, BSZ), 256, 0, stream>>>(g, W3, b3, out);
}
